// MultiEmbodimentActionEncoder_4741643895623
// MI455X (gfx1250) — matrix operand order repaired and checked by reading, not compiled, not run on hardware
//
#include <hip/hip_runtime.h>
#include <math.h>

#define E_ 32
#define A_ 64
#define H_ 1024
#define B_ 64
#define T_ 64

typedef __attribute__((ext_vector_type(16))) __bf16 v16bf;
typedef __attribute__((ext_vector_type(8)))  float  v8f;

union Frag {
    unsigned int u[8];
    uint4 q[2];
    v16bf v;
};

static __device__ __forceinline__ unsigned int pack2(float lo, float hi) {
    union { __bf16 h[2]; unsigned int u; } p;
    p.h[0] = (__bf16)lo;
    p.h[1] = (__bf16)hi;
    return p.u;
}

// Y[b, 0:64, nbase:nbase+128] = X[b, 0:64, 0:K] @ W[e] (K x 1024 row-major) + bias[e]
// optional swish. One workgroup per (n-tile, batch). 256 threads = 8 waves.
__global__ __launch_bounds__(256)
void cat_gemm_bf16(const float* __restrict__ X, int ldx,
                   const float* __restrict__ Wbase, long wstride_e,
                   const float* __restrict__ bias_base,
                   float* __restrict__ Y, int ldy,
                   int K, const int* __restrict__ cat_ids, int do_swish)
{
    // packed bf16 k-pairs: As[m][kp] holds (k=2kp, 2kp+1) of row m
    __shared__ __align__(16) unsigned int As[64][16];   // 4 KB
    // Ws[n][kp] holds (k=2kp, 2kp+1) of column n (transposed during staging)
    __shared__ __align__(16) unsigned int Ws[128][16];  // 8 KB

    const int b      = blockIdx.y;
    const int nbase  = blockIdx.x * 128;
    const int e      = cat_ids[b];
    const float* Wp  = Wbase + (long)e * wstride_e;     // K x H_ row-major
    const float* bp  = bias_base + (long)e * H_;
    const float* Xp  = X + (size_t)b * T_ * ldx;
    float*       Yp  = Y + (size_t)b * T_ * ldy;

    const int tid    = threadIdx.x;
    const int lane   = tid & 31;       // wave32
    const int wv     = tid >> 5;       // 0..7
    const int lane16 = lane & 15;
    const int laneHi = lane >> 4;

    v8f acc[4] = {};                   // 4 M-tiles x (16x16 f32) accumulators

    // staging assignments
    const int ar  = tid >> 2;          // 0..63  : A row
    const int ac  = (tid & 3) * 8;     // 0/8/16/24 : A col group (8 floats)
    const int wkp = tid >> 4;          // 0..15  : W k-pair
    const int wc  = (tid & 15) * 8;    // 0..120 : W col group (8 floats)

    for (int k0 = 0; k0 < K; k0 += 32) {
        // ---- stage A tile: 64 x 32 fp32 -> packed bf16 pairs ----
        const float* xr = Xp + (size_t)ar * ldx + k0 + ac;
        float4 x0 = *(const float4*)(xr);
        float4 x1 = *(const float4*)(xr + 4);
        As[ar][(ac >> 1) + 0] = pack2(x0.x, x0.y);
        As[ar][(ac >> 1) + 1] = pack2(x0.z, x0.w);
        As[ar][(ac >> 1) + 2] = pack2(x1.x, x1.y);
        As[ar][(ac >> 1) + 3] = pack2(x1.z, x1.w);

        // ---- stage W tile: 32 x 128 fp32, transposed + k-pair packed ----
        const float* wr0 = Wp + (size_t)(k0 + 2 * wkp) * H_ + nbase + wc;
        const float* wr1 = wr0 + H_;
        float4 w00 = *(const float4*)(wr0);
        float4 w01 = *(const float4*)(wr0 + 4);
        float4 w10 = *(const float4*)(wr1);
        float4 w11 = *(const float4*)(wr1 + 4);
        Ws[wc + 0][wkp] = pack2(w00.x, w10.x);
        Ws[wc + 1][wkp] = pack2(w00.y, w10.y);
        Ws[wc + 2][wkp] = pack2(w00.z, w10.z);
        Ws[wc + 3][wkp] = pack2(w00.w, w10.w);
        Ws[wc + 4][wkp] = pack2(w01.x, w11.x);
        Ws[wc + 5][wkp] = pack2(w01.y, w11.y);
        Ws[wc + 6][wkp] = pack2(w01.z, w11.z);
        Ws[wc + 7][wkp] = pack2(w01.w, w11.w);
        __syncthreads();

        // ---- B fragment (32x16, this wave's N strip) ----
        Frag bf;
        const unsigned int* brow = &Ws[wv * 16 + lane16][0];
        bf.q[0] = *(const uint4*)(brow + laneHi * 4);
        bf.q[1] = *(const uint4*)(brow + 8 + laneHi * 4);

        // ---- 4 M-tiles: A fragment + WMMA ----
        #pragma unroll
        for (int mt = 0; mt < 4; ++mt) {
            Frag af;
            const unsigned int* arow = &As[mt * 16 + lane16][0];
            af.q[0] = *(const uint4*)(arow + laneHi * 4);
            af.q[1] = *(const uint4*)(arow + 8 + laneHi * 4);
            acc[mt] = __builtin_amdgcn_wmma_f32_16x16x32_bf16(
                false, af.v, false, bf.v, (short)0, acc[mt], false, false);
        }
        __syncthreads();
    }

    // ---- epilogue: bias (+ swish) + store ----
    const int n = nbase + wv * 16 + lane16;
    const float bv = bp[n];
    #pragma unroll
    for (int mt = 0; mt < 4; ++mt) {
        const int trow = mt * 16 + laneHi * 8;   // C/D layout: lanes16-31 hold M=8..15
        #pragma unroll
        for (int r = 0; r < 8; ++r) {
            float v = acc[mt][r] + bv;
            if (do_swish) v = v / (1.0f + expf(-v));   // x * sigmoid(x)
            Yp[(size_t)(trow + r) * ldy + n] = v;
        }
    }
}

// Fill sinusoidal PE into xbuf[:, :, H:2H]; timestep is constant over T per batch.
__global__ __launch_bounds__(256)
void pe_fill(const int* __restrict__ timesteps, float* __restrict__ xbuf)
{
    const int b = blockIdx.y;
    const int t = blockIdx.x;
    const float ts = (float)timesteps[b];
    float* row = xbuf + ((size_t)b * T_ + t) * (2 * H_) + H_;
    const float c = logf(10000.0f) / 512.0f;   // ln(1e4)/half, half = 512
    #pragma unroll
    for (int i = 0; i < 4; ++i) {
        const int d  = threadIdx.x + 256 * i;
        const int dd = (d < 512) ? d : (d - 512);
        const float f = ts * expf(-(float)dd * c);
        row[d] = (d < 512) ? sinf(f) : cosf(f);
    }
}

extern "C" void kernel_launch(void* const* d_in, const int* in_sizes, int n_in,
                              void* d_out, int out_size, void* d_ws, size_t ws_size,
                              hipStream_t stream)
{
    const float* actions = (const float*)d_in[0];   // (B,T,A)
    const float* W1      = (const float*)d_in[1];   // (E,A,H)
    const float* b1      = (const float*)d_in[2];   // (E,H)
    const float* W2      = (const float*)d_in[3];   // (E,2H,H)
    const float* b2      = (const float*)d_in[4];   // (E,H)
    const float* W3      = (const float*)d_in[5];   // (E,H,H)
    const float* b3      = (const float*)d_in[6];   // (E,H)
    const int*   ts      = (const int*)d_in[7];     // (B,)
    const int*   cid     = (const int*)d_in[8];     // (B,)
    float*       out     = (float*)d_out;           // (B,T,H)

    float* xbuf = (float*)d_ws;                         // B*T*2H fp32 (32 MB)
    float* hbuf = xbuf + (size_t)B_ * T_ * 2 * H_;      // B*T*H  fp32 (16 MB)

    dim3 blk(256);
    dim3 grd(H_ / 128, B_);

    // Layer 1: actions @ W1[e] + b1 -> xbuf[:, :, 0:H]
    cat_gemm_bf16<<<grd, blk, 0, stream>>>(actions, A_, W1, (long)A_ * H_, b1,
                                           xbuf, 2 * H_, A_, cid, 0);
    // Sinusoidal PE -> xbuf[:, :, H:2H]
    pe_fill<<<dim3(T_, B_), blk, 0, stream>>>(ts, xbuf);
    // Layer 2 + swish: xbuf @ W2[e] + b2 -> hbuf
    cat_gemm_bf16<<<grd, blk, 0, stream>>>(xbuf, 2 * H_, W2, (long)2 * H_ * H_, b2,
                                           hbuf, H_, 2 * H_, cid, 1);
    // Layer 3: hbuf @ W3[e] + b3 -> out
    cat_gemm_bf16<<<grd, blk, 0, stream>>>(hbuf, H_, W3, (long)H_ * H_, b3,
                                           out, H_, H_, cid, 0);
}
